// Transformer_84877143704114
// MI455X (gfx1250) — matrix operand order repaired and checked by reading, not compiled, not run on hardware
//
#include <hip/hip_runtime.h>
#include <hip/hip_bf16.h>

typedef __attribute__((ext_vector_type(16))) _Float16 v16h;
typedef __attribute__((ext_vector_type(8)))  float    v8f;
typedef __attribute__((ext_vector_type(2)))  _Float16 h2v;
typedef __attribute__((ext_vector_type(4)))  unsigned v4u;
typedef __attribute__((ext_vector_type(4)))  int      v4i;
typedef __attribute__((ext_vector_type(8)))  int      v8i;

#define HEADS   8
#define DIMH    48
#define NTOK    1024
#define DMODEL  384
#define BATCH   16
#define ROWS    (BATCH * NTOK)          // 16384
#define HID     1536
#define QKVW    (3 * HEADS * DIMH)      // 1152
#define ATT_SCALE 0.14433756729740643f  // 48^-0.5

// ---------------------------------------------------------------------------
// fragment index helpers (CDNA5 wave32 WMMA layouts, ISA 7.12.2)
// ---------------------------------------------------------------------------
__device__ __forceinline__ int a_frag_kbase(int v, int g) {
    return ((v < 4) ? (v * 2) : (16 + (v - 4) * 2)) + g * 8;
}

// ---------------------------------------------------------------------------
// DPP16 butterfly reduction across the 16 lanes of each half-row (wave32).
// ---------------------------------------------------------------------------
template <int CTRL>
__device__ __forceinline__ float dpp_movf(float x) {
    return __builtin_bit_cast(float,
        __builtin_amdgcn_update_dpp(0, __builtin_bit_cast(int, x), CTRL, 0xF, 0xF, true));
}
__device__ __forceinline__ float row16_max(float x) {
    x = fmaxf(x, dpp_movf<0xB1>(x));    // quad_perm(1,0,3,2)
    x = fmaxf(x, dpp_movf<0x4E>(x));    // quad_perm(2,3,0,1)
    x = fmaxf(x, dpp_movf<0x141>(x));   // row_half_mirror
    x = fmaxf(x, dpp_movf<0x140>(x));   // row_mirror
    return x;
}
__device__ __forceinline__ float row16_sum(float x) {
    x += dpp_movf<0xB1>(x);
    x += dpp_movf<0x4E>(x);
    x += dpp_movf<0x141>(x);
    x += dpp_movf<0x140>(x);
    return x;
}

// ---------------------------------------------------------------------------
// CDNA5 async global -> LDS copy (16B per lane), tracked by ASYNCcnt.
// ---------------------------------------------------------------------------
__device__ __forceinline__ void async_copy_b128(const void* lds_ptr, const void* gptr) {
    unsigned           loff = (unsigned)(unsigned long long)lds_ptr;
    unsigned long long gadr = (unsigned long long)gptr;
    asm volatile("global_load_async_to_lds_b128 %0, %1, off"
                 :: "v"(loff), "v"(gadr) : "memory");
}
__device__ __forceinline__ void async_wait0() {
    asm volatile("s_wait_asynccnt 0" ::: "memory");
}
// wave-local LDS ordering fence (DS ops of a wave complete in order)
__device__ __forceinline__ void lds_wave_fence() {
    asm volatile("s_wait_dscnt 0" ::: "memory");
    __builtin_amdgcn_wave_barrier();
}

// ---------------------------------------------------------------------------
// TDM: 2D tile (nRows x rowBytes, row stride rowStrideBytes) -> LDS, with
// LDS padding pad_amount=4 DWORDs after every pad_interval=16 DWORDs
// (i.e. 64B data + 16B pad = 80B pitch).  Descriptor per ISA 8.3/8.4.
// Issue from ONE wave only (TDM ignores EXEC).  Tracked by TENSORcnt.
// ---------------------------------------------------------------------------
__device__ __forceinline__ void tdm_load_2d_pad(const void* lds_dst, const void* gsrc,
                                                int rowBytes, int nRows,
                                                long long rowStrideBytes) {
    unsigned long long ga = (unsigned long long)gsrc;
    v4u g0;
    g0[0] = 1u;                                            // count=1 (valid user D#)
    g0[1] = (unsigned)(unsigned long long)lds_dst;         // lds_addr (bytes)
    g0[2] = (unsigned)ga;                                  // global_addr[31:0]
    g0[3] = (unsigned)((ga >> 32) & 0x01FFFFFFull) | (2u << 30);  // addr[56:32] | type=2
    v8i g1;
    g1[0] = (1 << 20) | (3 << 22) | (3 << 25);             // pad_en, intv=16DW, amt=4DW
    g1[1] = 0;                                             // no atomic barrier; dim0 lo16=0
    g1[2] = 0x4000;                                        // tensor_dim0=1<<30 (no clip)
    g1[3] = 0x4000 | (rowBytes << 16);                     // tensor_dim1=1<<30; tile_dim0
    g1[4] = nRows & 0xFFFF;                                // tile_dim1 (tile_dim2=0)
    g1[5] = (int)(rowStrideBytes & 0xFFFFFFFFll);          // dim0_stride[31:0]
    g1[6] = (int)((rowStrideBytes >> 32) & 0xFFFFll);      // dim0_stride[47:32]
    g1[7] = 0;                                             // dim1_stride=0
    v4i z4 = {0, 0, 0, 0};
#if __clang_major__ >= 23
    v8i z8 = {0, 0, 0, 0, 0, 0, 0, 0};
    __builtin_amdgcn_tensor_load_to_lds(g0, g1, z4, z4, z8, 0);
#else
    __builtin_amdgcn_tensor_load_to_lds(g0, g1, z4, z4, 0);
#endif
}
__device__ __forceinline__ void tensor_wait0() {
    __builtin_amdgcn_s_wait_tensorcnt((short)0);
}

// ---------------------------------------------------------------------------
// elementwise f32 -> f16
// ---------------------------------------------------------------------------
__global__ __launch_bounds__(256) void cvt_f32_f16(const float* __restrict__ in,
                                                   _Float16* __restrict__ out, int n) {
    int i = blockIdx.x * 256 + threadIdx.x;
    if (i < n) out[i] = (_Float16)in[i];
}

// ---------------------------------------------------------------------------
// LayerNorm over last dim (384), f32 in -> f16 out.  1 block (128 thr) per row.
// ---------------------------------------------------------------------------
__global__ __launch_bounds__(128) void ln_kernel(const float* __restrict__ X,
                                                 const float* __restrict__ w,
                                                 const float* __restrict__ b,
                                                 _Float16* __restrict__ out) {
    int row  = blockIdx.x;
    int tid  = threadIdx.x;
    int lane = tid & 31, wv = tid >> 5;
    const float* xr = X + (size_t)row * DMODEL;

    float s = 0.f, s2 = 0.f;
    #pragma unroll
    for (int c = tid; c < DMODEL; c += 128) {
        float v = xr[c];
        s += v; s2 += v * v;
    }
    s  = row16_sum(s);
    s2 = row16_sum(s2);
    __shared__ float sh[16];
    if ((lane & 15) == 0) {
        int slot = wv * 2 + (lane >> 4);
        sh[slot] = s; sh[8 + slot] = s2;
    }
    __syncthreads();
    s  = sh[0] + sh[1] + sh[2] + sh[3] + sh[4] + sh[5] + sh[6] + sh[7];
    s2 = sh[8] + sh[9] + sh[10] + sh[11] + sh[12] + sh[13] + sh[14] + sh[15];
    float mu  = s * (1.f / DMODEL);
    float var = s2 * (1.f / DMODEL) - mu * mu;
    float r   = rsqrtf(var + 1e-5f);
    #pragma unroll
    for (int c = tid; c < DMODEL; c += 128)
        out[(size_t)row * DMODEL + c] = (_Float16)((xr[c] - mu) * r * w[c] + b[c]);
}

// ---------------------------------------------------------------------------
// WMMA GEMM:  C[M,N] = A[M,K] (f16) * B[K,N] (f16), f32 accumulate.
// 256 thr = 8 waves; block tile 64x128, BK=32; wave tile 16x64.
// Double-buffered LDS; A tile via TDM (tensor_load_to_lds, padded pitch);
// B tile paired-row transpose (b32 stores).  One block barrier per k-step.
// ---------------------------------------------------------------------------
#define EPI_F16       0
#define EPI_BIAS_RES  1
#define EPI_BIAS_GELU 2

template <int EPI>
__global__ __launch_bounds__(256) void gemm_wmma(const _Float16* __restrict__ A,
                                                 const _Float16* __restrict__ B,
                                                 const float* __restrict__ bias,
                                                 const float* __restrict__ resid,
                                                 void* __restrict__ outv,
                                                 int M, int K, int N) {
    __shared__ _Float16 As[2][64 * 40];   // 64 rows x (32+8 pad) = 80B pitch (TDM pad)
    __shared__ _Float16 Bs[2][128 * 40];  // 128 n-cols x (32+8 pad), k-contig

    int tid  = threadIdx.x;
    int lane = tid & 31, wv = tid >> 5;
    int nBlkN = N >> 7;
    int m0 = (blockIdx.x / nBlkN) * 64;
    int n0 = (blockIdx.x % nBlkN) * 128;
    int wm = wv & 3, wn = wv >> 2;
    int g = lane >> 4, nl = lane & 15;

    auto stageA = [&](int buf, int k0) {      // one TDM op per block
        if (wv == 0)
            tdm_load_2d_pad(&As[buf][0], A + (size_t)m0 * K + k0,
                            64, 64, (long long)K * 2);
    };
    auto stageB = [&](int buf, int k0) {      // row pairs -> h2v stores
        int r = (tid >> 4) * 2, cc = (tid & 15) * 8;
        const _Float16* s0 = B + (size_t)(k0 + r) * N + n0 + cc;
        uint4 p0 = *(const uint4*)s0;
        uint4 p1 = *(const uint4*)(s0 + N);
        const _Float16* h0 = (const _Float16*)&p0;
        const _Float16* h1 = (const _Float16*)&p1;
        #pragma unroll
        for (int i = 0; i < 8; i++) {
            h2v pr; pr[0] = h0[i]; pr[1] = h1[i];
            *(h2v*)(&Bs[buf][(cc + i) * 40 + r]) = pr;
        }
    };

    v8f zero8 = {0.f, 0.f, 0.f, 0.f, 0.f, 0.f, 0.f, 0.f};
    v8f acc[4];
    #pragma unroll
    for (int j = 0; j < 4; j++) acc[j] = zero8;

    stageA(0, 0);
    stageB(0, 0);
    tensor_wait0();
    __syncthreads();

    for (int k0 = 0; k0 < K; k0 += 32) {
        int cur = (k0 >> 5) & 1;
        if (k0 + 32 < K) {                 // prefetch next tile (uniform predicate)
            stageA(cur ^ 1, k0 + 32);
            stageB(cur ^ 1, k0 + 32);
        }

        v16h af;
        int mrow = wm * 16 + nl;
        #pragma unroll
        for (int v = 0; v < 8; v++) {
            h2v p = *(const h2v*)(&As[cur][mrow * 40 + a_frag_kbase(v, g)]);
            af[2 * v] = p[0]; af[2 * v + 1] = p[1];
        }
        #pragma unroll
        for (int j = 0; j < 4; j++) {
            int ncol = wn * 64 + j * 16 + nl;
            v16h bf;
            #pragma unroll
            for (int v = 0; v < 8; v++) {
                h2v p = *(const h2v*)(&Bs[cur][ncol * 40 + a_frag_kbase(v, g)]);
                bf[2 * v] = p[0]; bf[2 * v + 1] = p[1];
            }
            acc[j] = __builtin_amdgcn_wmma_f32_16x16x32_f16(
                false, af, false, bf, (short)0, acc[j], false, false);
        }
        tensor_wait0();                    // issuing wave drains TDM; others no-op
        __syncthreads();
    }

    #pragma unroll
    for (int j = 0; j < 4; j++) {
        #pragma unroll
        for (int v = 0; v < 8; v++) {
            int row = m0 + wm * 16 + v + 8 * g;
            int col = n0 + wn * 64 + j * 16 + nl;
            float val = acc[j][v];
            if constexpr (EPI == EPI_F16) {
                ((_Float16*)outv)[(size_t)row * N + col] = (_Float16)val;
            } else if constexpr (EPI == EPI_BIAS_RES) {
                val += bias[col] + resid[(size_t)row * N + col];
                ((float*)outv)[(size_t)row * N + col] = val;
            } else {  // exact erf GELU
                val += bias[col];
                val = 0.5f * val * (1.f + erff(val * 0.70710678118654752f));
                ((_Float16*)outv)[(size_t)row * N + col] = (_Float16)val;
            }
        }
    }
}

// ---------------------------------------------------------------------------
// Attention: block = 4 waves, all same (b, head); wave w handles q-tile qt0+w.
// Double-buffered K (async-to-LDS, pad pre-zeroed) and V^T (paired-row b32
// transpose). DPP16 softmax stats; LDS bias LUT; one block barrier per step.
// ---------------------------------------------------------------------------
__global__ __launch_bounds__(128) void attn_kernel(const _Float16* __restrict__ qkv,
                                                   const float* __restrict__ bias_table,
                                                   _Float16* __restrict__ attn_out) {
    __shared__ float    biasT[63 * 63];
    __shared__ _Float16 Ks[2][32 * 64];   // [key][d], d padded to 64 (zeros)
    __shared__ _Float16 Vs[2][48 * 32];   // [d][key] (transposed)
    __shared__ _Float16 ldsP[4][16 * 32];

    int tid = threadIdx.x, lane = tid & 31, w = tid >> 5;
    int wid  = blockIdx.x * 4 + w;
    int b    = wid >> 9;
    int head = (wid >> 6) & 7;
    int qt   = wid & 63;
    int g = lane >> 4, nl = lane & 15;
    int qbase = qt * 16;

    // per-head relative-position bias LUT
    for (int i = tid; i < 63 * 63; i += 128) {
        int dy = i / 63 - 31, dx = i % 63 - 31;
        int r2 = dy * dy + dx * dx;
        int s = (int)__builtin_amdgcn_sqrtf((float)r2);
        s += (s * s < r2) ? 1 : 0;
        biasT[i] = bias_table[s * HEADS + head];
    }
    // pre-zero K pad columns [row][48..63] in both buffers (never overwritten)
    {
        uint4 z = {0u, 0u, 0u, 0u};
        int i = tid;                    // 2 bufs x 32 rows x 2 chunks = 128
        int bufi = i >> 6, row = (i >> 1) & 31, ch = i & 1;
        *(uint4*)(&Ks[bufi][row * 64 + 48 + ch * 8]) = z;
    }

    auto stageK = [&](int buf, int kb) {
        for (int idx = tid; idx < 192; idx += 128) {   // 32 rows x 6 chunks
            int row = idx / 6, ch = idx % 6;
            async_copy_b128(&Ks[buf][row * 64 + ch * 8],
                            qkv + (size_t)(b * NTOK + kb + row) * QKVW +
                                DMODEL + head * DIMH + ch * 8);
        }
    };
    auto stageV = [&](int buf, int kb) {               // 16 row-pairs x 6 chunks
        for (int idx = tid; idx < 96; idx += 128) {
            int rp = idx / 6, ch = idx % 6;
            int r = rp * 2;
            const _Float16* s0 = qkv + (size_t)(b * NTOK + kb + r) * QKVW +
                                 2 * DMODEL + head * DIMH + ch * 8;
            uint4 p0 = *(const uint4*)s0;
            uint4 p1 = *(const uint4*)(s0 + QKVW);
            const _Float16* h0 = (const _Float16*)&p0;
            const _Float16* h1 = (const _Float16*)&p1;
            #pragma unroll
            for (int i = 0; i < 8; i++) {
                h2v pr; pr[0] = h0[i]; pr[1] = h1[i];
                *(h2v*)(&Vs[buf][(ch * 8 + i) * 32 + r]) = pr;
            }
        }
    };

    // Q A-fragments (scale folded in), d padded 48 -> 64
    const _Float16* qrow = qkv + (size_t)(b * NTOK + qbase + nl) * QKVW + head * DIMH;
    v16h qA[2];
    #pragma unroll
    for (int c = 0; c < 2; c++) {
        #pragma unroll
        for (int e = 0; e < 16; e++) {
            int v = e >> 1, h = e & 1;
            int kd = a_frag_kbase(v, g) + h + c * 32;
            float qv = (kd < DIMH) ? (float)qrow[kd] : 0.f;
            qA[c][e] = (_Float16)(qv * ATT_SCALE);
        }
    }

    v8f zero8 = {0.f, 0.f, 0.f, 0.f, 0.f, 0.f, 0.f, 0.f};
    v8f O[3];
    #pragma unroll
    for (int j = 0; j < 3; j++) O[j] = zero8;
    float mRow[8], lRow[8];
    int qy[8], qx[8];
    #pragma unroll
    for (int v = 0; v < 8; v++) {
        mRow[v] = -1e30f; lRow[v] = 0.f;
        int qg = qbase + v + 8 * g;
        qy[v] = qg >> 5; qx[v] = qg & 31;
    }

    stageK(0, 0);
    stageV(0, 0);
    async_wait0();
    __syncthreads();   // also publishes LUT + pad zeros

    for (int kb = 0; kb < NTOK; kb += 32) {
        int cur = (kb >> 5) & 1;
        if (kb + 32 < NTOK) {              // prefetch next K/V tile
            stageK(cur ^ 1, kb + 32);
            stageV(cur ^ 1, kb + 32);
        }
        const _Float16* KsC = Ks[cur];
        const _Float16* VsC = Vs[cur];

        // ---- S = scale*Q.K^T + rel_bias ----
        v8f S[2];
        #pragma unroll
        for (int t = 0; t < 2; t++) {
            v8f s = zero8;
            #pragma unroll
            for (int c = 0; c < 2; c++) {
                v16h kB;
                #pragma unroll
                for (int v = 0; v < 8; v++) {
                    h2v p = *(const h2v*)(KsC + (t * 16 + nl) * 64 + c * 32 + a_frag_kbase(v, g));
                    kB[2 * v] = p[0]; kB[2 * v + 1] = p[1];
                }
                s = __builtin_amdgcn_wmma_f32_16x16x32_f16(
                    false, qA[c], false, kB, (short)0, s, false, false);
            }
            int kg = kb + t * 16 + nl;
            int ky = kg >> 5, kx = kg & 31;
            #pragma unroll
            for (int v = 0; v < 8; v++)
                s[v] += biasT[(qy[v] - ky + 31) * 63 + (qx[v] - kx + 31)];
            S[t] = s;
        }

        // ---- online softmax statistics (DPP16) ----
        float mNew[8], alpha[8];
        #pragma unroll
        for (int v = 0; v < 8; v++) {
            float t = row16_max(fmaxf(S[0][v], S[1][v]));
            mNew[v]  = fmaxf(mRow[v], t);
            alpha[v] = __expf(mRow[v] - mNew[v]);
        }
        #pragma unroll
        for (int v = 0; v < 8; v++) {
            S[0][v] = __expf(S[0][v] - mNew[v]);
            S[1][v] = __expf(S[1][v] - mNew[v]);
            float sm = row16_sum(S[0][v] + S[1][v]);
            lRow[v] = lRow[v] * alpha[v] + sm;
            mRow[v] = mNew[v];
        }
        #pragma unroll
        for (int j = 0; j < 3; j++)
            #pragma unroll
            for (int v = 0; v < 8; v++) O[j][v] *= alpha[v];

        // ---- C-layout P -> LDS -> A-layout fragment (wave-local fence) ----
        #pragma unroll
        for (int t = 0; t < 2; t++)
            #pragma unroll
            for (int v = 0; v < 8; v++)
                ldsP[w][(v + 8 * g) * 32 + t * 16 + nl] = (_Float16)S[t][v];
        lds_wave_fence();
        v16h pA;
        #pragma unroll
        for (int v = 0; v < 8; v++) {
            h2v p = *(const h2v*)&ldsP[w][nl * 32 + a_frag_kbase(v, g)];
            pA[2 * v] = p[0]; pA[2 * v + 1] = p[1];
        }

        // ---- O += P . V ----
        #pragma unroll
        for (int j = 0; j < 3; j++) {
            v16h vB;
            #pragma unroll
            for (int v = 0; v < 8; v++) {
                h2v p = *(const h2v*)(VsC + (j * 16 + nl) * 32 + a_frag_kbase(v, g));
                vB[2 * v] = p[0]; vB[2 * v + 1] = p[1];
            }
            O[j] = __builtin_amdgcn_wmma_f32_16x16x32_f16(
                false, pA, false, vB, (short)0, O[j], false, false);
        }

        async_wait0();
        __syncthreads();   // next-tile staging complete; current reads done
    }

    // ---- normalize (rcp) and write [b, n, head*48 + d] as f16 ----
    float linv[8];
    #pragma unroll
    for (int v = 0; v < 8; v++) linv[v] = __builtin_amdgcn_rcpf(lRow[v]);
    #pragma unroll
    for (int j = 0; j < 3; j++)
        #pragma unroll
        for (int v = 0; v < 8; v++) {
            int row = b * NTOK + qbase + v + 8 * g;
            attn_out[(size_t)row * DMODEL + head * DIMH + j * 16 + nl] =
                (_Float16)(O[j][v] * linv[v]);
        }
}

// ---------------------------------------------------------------------------
// launcher
// ---------------------------------------------------------------------------
extern "C" void kernel_launch(void* const* d_in, const int* in_sizes, int n_in,
                              void* d_out, int out_size, void* d_ws, size_t ws_size,
                              hipStream_t stream) {
    const float* x          = (const float*)d_in[0];
    const float* ln1_w      = (const float*)d_in[1];
    const float* ln1_b      = (const float*)d_in[2];
    const float* w_qkv      = (const float*)d_in[3];
    const float* bias_table = (const float*)d_in[4];
    const float* w_out      = (const float*)d_in[5];
    const float* b_out      = (const float*)d_in[6];
    const float* ln2_w      = (const float*)d_in[7];
    const float* ln2_b      = (const float*)d_in[8];
    const float* w1         = (const float*)d_in[9];
    const float* b1         = (const float*)d_in[10];
    const float* w2         = (const float*)d_in[11];
    const float* b2         = (const float*)d_in[12];
    float* out              = (float*)d_out;

    char* p = (char*)d_ws;
    auto carve = [&](size_t bytes) -> void* {
        void* r = (void*)p;
        p += (bytes + 255) & ~(size_t)255;
        return r;
    };
    _Float16* wqkv_h = (_Float16*)carve((size_t)DMODEL * QKVW * 2);
    _Float16* wout_h = (_Float16*)carve((size_t)DMODEL * DMODEL * 2);
    _Float16* w1_h   = (_Float16*)carve((size_t)DMODEL * HID * 2);
    _Float16* w2_h   = (_Float16*)carve((size_t)HID * DMODEL * 2);
    _Float16* h16    = (_Float16*)carve((size_t)ROWS * DMODEL * 2);
    _Float16* qkv_h  = (_Float16*)carve((size_t)ROWS * QKVW * 2);
    _Float16* attn_h = (_Float16*)carve((size_t)ROWS * DMODEL * 2);
    float*    xt1    = (float*)   carve((size_t)ROWS * DMODEL * 4);
    _Float16* ffn1_h = (_Float16*)carve((size_t)ROWS * HID * 2);

    auto cvt = [&](const float* src, _Float16* dst, int n) {
        cvt_f32_f16<<<(n + 255) / 256, 256, 0, stream>>>(src, dst, n);
    };
    cvt(w_qkv, wqkv_h, DMODEL * QKVW);
    cvt(w_out, wout_h, DMODEL * DMODEL);
    cvt(w1,    w1_h,   DMODEL * HID);
    cvt(w2,    w2_h,   HID * DMODEL);

    ln_kernel<<<ROWS, 128, 0, stream>>>(x, ln1_w, ln1_b, h16);

    gemm_wmma<EPI_F16><<<(ROWS / 64) * (QKVW / 128), 256, 0, stream>>>(
        h16, wqkv_h, nullptr, nullptr, qkv_h, ROWS, DMODEL, QKVW);

    attn_kernel<<<BATCH * HEADS * 64 / 4, 128, 0, stream>>>(qkv_h, bias_table, attn_h);

    gemm_wmma<EPI_BIAS_RES><<<(ROWS / 64) * (DMODEL / 128), 256, 0, stream>>>(
        attn_h, wout_h, b_out, x, xt1, ROWS, DMODEL, DMODEL);

    ln_kernel<<<ROWS, 128, 0, stream>>>(xt1, ln2_w, ln2_b, h16);

    gemm_wmma<EPI_BIAS_GELU><<<(ROWS / 64) * (HID / 128), 256, 0, stream>>>(
        h16, w1_h, b1, nullptr, ffn1_h, ROWS, DMODEL, HID);

    gemm_wmma<EPI_BIAS_RES><<<(ROWS / 64) * (DMODEL / 128), 256, 0, stream>>>(
        ffn1_h, w2_h, b2, xt1, out, ROWS, HID, DMODEL);
}
